// GCN_44358422233706
// MI455X (gfx1250) — matrix operand order repaired and checked by reading, not compiled, not run on hardware
//
#include <hip/hip_runtime.h>

#define M_TOT   96
#define N_ATOMS 96
#define N_BONDS 96
#define MAX_NB  3
#define HB      158
#define HB2     316
#define KPAD    320
#define BM      4          // molecules batched into WMMA M dimension per workgroup
#define NW      10         // waves = N-tiles of 16 covering 158 outputs
#define THREADS (NW * 32)
#define STEPS   (N_BONDS * MAX_NB)
#define FB_STRIDE 160      // padded row stride (16B aligned)

typedef __attribute__((ext_vector_type(16))) __bf16 v16bf;
typedef __attribute__((ext_vector_type(8)))  float  v8f;

// f32 -> bf16, round-to-nearest-even
__device__ __forceinline__ unsigned short f2bf(float f) {
  unsigned u = __float_as_uint(f);
  u += 0x7FFFu + ((u >> 16) & 1u);
  return (unsigned short)(u >> 16);
}

// LDS layout (bytes), total 266240 <= 320KB WGP LDS
#define OFF_FB   0          // float fb[BM][96][160]      : 245760 B
#define OFF_BS   245760     // float bs[BM][160]          : 2560 B
#define OFF_X    248320     // ushort X[16][320] (bf16)   : 10240 B (16B aligned)
#define OFF_A2B  258560     // int  a2b[BM][96][3]        : 4608 B
#define OFF_B2A  263168     // int  b2a[BM][96]           : 1536 B
#define OFF_RB   264704     // int  b2revb[BM][96]        : 1536 B
#define LDS_BYTES 266240

__global__ __launch_bounds__(THREADS)
void mpnn_scan_kernel(const int* __restrict__ a2b, const int* __restrict__ b2a,
                      const int* __restrict__ b2revb,
                      const float* __restrict__ f_bonds,
                      const float* __restrict__ bond_sum,
                      const float* __restrict__ W0, const float* __restrict__ b0,
                      float* __restrict__ mp_out) {
  extern __shared__ char smem_raw[];
  float*          fb   = (float*)(smem_raw + OFF_FB);
  float*          bs   = (float*)(smem_raw + OFF_BS);
  unsigned short* Xs   = (unsigned short*)(smem_raw + OFF_X);
  int*            sA2B = (int*)(smem_raw + OFF_A2B);
  int*            sB2A = (int*)(smem_raw + OFF_B2A);
  int*            sRB  = (int*)(smem_raw + OFF_RB);

  const int tid  = threadIdx.x;
  const int lane = tid & 31;
  const int w    = tid >> 5;          // this wave's N-tile
  const int mol0 = blockIdx.x * BM;

  // ---- Stage W0 slice for this wave's 16 outputs in VGPRs as bf16 B-tiles
  // B tile (32x16) layout: lanes 0-15 -> N=lane, K=2e,2e+1; lanes 16-31 -> K+16.
  v16bf Breg[10];
  {
    const int n = 16 * w + (lane & 15);
    #pragma unroll
    for (int kt = 0; kt < 10; ++kt) {
      union { unsigned u[8]; v16bf v; } bb;
      #pragma unroll
      for (int e = 0; e < 8; ++e) {
        const int k = 32 * kt + ((lane >> 4) << 4) + 2 * e;
        unsigned lo = 0, hi = 0;
        if (n < HB) {
          if (k     < HB2) lo = (unsigned)f2bf(W0[n * HB2 + k]);
          if (k + 1 < HB2) hi = (unsigned)f2bf(W0[n * HB2 + k + 1]);
        }
        bb.u[e] = lo | (hi << 16);
      }
      Breg[kt] = bb.v;
    }
  }
  const int   n_me  = 16 * w + (lane & 15);
  const float biasv = (n_me < HB) ? b0[n_me] : 0.0f;

  // ---- Load state + index tables into LDS
  for (int idx = tid; idx < BM * N_BONDS * HB; idx += THREADS) {
    int m = idx / (N_BONDS * HB);
    int r = idx - m * (N_BONDS * HB);
    int c = r / HB;
    int i = r - c * HB;
    fb[(m * N_BONDS + c) * FB_STRIDE + i] =
        f_bonds[((size_t)(mol0 + m) * N_BONDS + c) * HB + i];
  }
  for (int idx = tid; idx < BM * HB; idx += THREADS) {
    int m = idx / HB, i = idx - m * HB;
    bs[m * FB_STRIDE + i] = bond_sum[(mol0 + m) * HB + i];
  }
  for (int idx = tid; idx < 16 * KPAD; idx += THREADS) Xs[idx] = 0;  // dead rows/K-pad = 0
  for (int idx = tid; idx < BM * N_ATOMS * MAX_NB; idx += THREADS) {
    int m = idx / (N_ATOMS * MAX_NB);
    sA2B[idx] = a2b[(mol0 + m) * N_ATOMS * MAX_NB + (idx - m * N_ATOMS * MAX_NB)];
  }
  for (int idx = tid; idx < BM * N_BONDS; idx += THREADS) {
    int m = idx / N_BONDS, r = idx - m * N_BONDS;
    sB2A[idx] = b2a[(mol0 + m) * N_BONDS + r];
    sRB[idx]  = b2revb[(mol0 + m) * N_BONDS + r];
  }
  __syncthreads();

  // A tile (16x32 bf16) layout: lane -> M=lane%16, half h=lane/16:
// VGPRs 0-3 = K [16h..16h+7], VGPRs 4-7 = K [16h+8..16h+15]  => two b128 LDS loads
  const unsigned short* xrow = Xs + (lane & 15) * KPAD;
  const int h8 = (lane >> 4) << 3;

  for (int t = 0; t < STEPS; ++t) {
    const int c = t / MAX_NB;
    const int j = t - c * MAX_NB;

    // ---- prep: carried bond_sum update + bf16 staging of X = [fb[c] | bs]
    for (int idx = tid; idx < BM * HB; idx += THREADS) {
      int m = idx / HB, i = idx - m * HB;
      int atom = sB2A[m * N_BONDS + c];
      int d    = sA2B[(m * N_ATOMS + atom) * MAX_NB + j];
      int rb   = sRB[m * N_BONDS + c];
      float fcb = fb[(m * N_BONDS + c) * FB_STRIDE + i];
      float v   = bs[m * FB_STRIDE + i]
                + fb[(m * N_BONDS + (d - 1)) * FB_STRIDE + i]
                - fb[(m * N_BONDS + rb) * FB_STRIDE + i];
      bs[m * FB_STRIDE + i] = v;
      Xs[m * KPAD + i]      = f2bf(fcb);
      Xs[m * KPAD + HB + i] = f2bf(v);
    }
    __syncthreads();

    // ---- NF(16x16 tile) = X(16x320) @ W0^T slice + b0 on the matrix pipes
    v8f acc;
    #pragma unroll
    for (int r = 0; r < 8; ++r) acc[r] = biasv;
    #pragma unroll
    for (int kt = 0; kt < 10; ++kt) {
      union { int4 q[2]; v16bf v; } aa;
      aa.q[0] = *(const int4*)(xrow + 32 * kt + 2 * h8);
      aa.q[1] = *(const int4*)(xrow + 32 * kt + 8 + 2 * h8);
      acc = __builtin_amdgcn_wmma_f32_16x16x32_bf16(
          false, aa.v, false, Breg[kt], (short)0, acc, false, false);
    }
    // write back fb[c] for the BM live molecules (lanes 0-15 hold M = r)
    if ((lane < 16) && (n_me < HB)) {
      #pragma unroll
      for (int r = 0; r < BM; ++r)
        fb[(r * N_BONDS + c) * FB_STRIDE + n_me] = acc[r];
    }
    __syncthreads();
  }

  // ---- mp = fb.sum(over bonds)
  for (int idx = tid; idx < BM * HB; idx += THREADS) {
    int m = idx / HB, i = idx - m * HB;
    float s = 0.f;
    for (int c = 0; c < N_BONDS; ++c) s += fb[(m * N_BONDS + c) * FB_STRIDE + i];
    mp_out[(mol0 + m) * HB + i] = s;
  }
}

// ---- tiny MLP head: 158 -> 110 -> relu(70) -> relu(35) -> 1, one block per molecule
__global__ __launch_bounds__(128)
void mlp_head_kernel(const float* __restrict__ mp,
                     const float* __restrict__ W1, const float* __restrict__ b1,
                     const float* __restrict__ W2, const float* __restrict__ b2,
                     const float* __restrict__ W3, const float* __restrict__ b3,
                     const float* __restrict__ W4, const float* __restrict__ b4,
                     float* __restrict__ out) {
  __shared__ float s_mp[HB];
  __shared__ float s_h1[110];
  __shared__ float s_h2[70];
  __shared__ float s_h3[35];
  const int m = blockIdx.x, tid = threadIdx.x;
  for (int i = tid; i < HB; i += 128) s_mp[i] = mp[m * HB + i];
  __syncthreads();
  if (tid < 110) {               // layer 1: NO activation (per reference)
    float a = b1[tid];
    for (int k = 0; k < HB; ++k) a += s_mp[k] * W1[tid * HB + k];
    s_h1[tid] = a;
  }
  __syncthreads();
  if (tid < 70) {
    float a = b2[tid];
    for (int k = 0; k < 110; ++k) a += s_h1[k] * W2[tid * 110 + k];
    s_h2[tid] = fmaxf(a, 0.f);
  }
  __syncthreads();
  if (tid < 35) {
    float a = b3[tid];
    for (int k = 0; k < 70; ++k) a += s_h2[k] * W3[tid * 70 + k];
    s_h3[tid] = fmaxf(a, 0.f);
  }
  __syncthreads();
  if (tid == 0) {
    float a = b4[0];
    for (int k = 0; k < 35; ++k) a += s_h3[k] * W4[k];
    out[m] = a;
  }
}

extern "C" void kernel_launch(void* const* d_in, const int* in_sizes, int n_in,
                              void* d_out, int out_size, void* d_ws, size_t ws_size,
                              hipStream_t stream) {
  (void)in_sizes; (void)n_in; (void)out_size; (void)ws_size;
  const int*   a2b      = (const int*)d_in[0];
  const int*   b2a      = (const int*)d_in[1];
  const int*   b2revb   = (const int*)d_in[2];
  const float* f_bonds  = (const float*)d_in[3];
  /* d_in[4] = f_atoms_ is unused by the reference forward */
  const float* bond_sum = (const float*)d_in[5];
  const float* W0 = (const float*)d_in[6];
  const float* b0 = (const float*)d_in[7];
  const float* W1 = (const float*)d_in[8];
  const float* b1 = (const float*)d_in[9];
  const float* W2 = (const float*)d_in[10];
  const float* b2 = (const float*)d_in[11];
  const float* W3 = (const float*)d_in[12];
  const float* b3 = (const float*)d_in[13];
  const float* W4 = (const float*)d_in[14];
  const float* b4 = (const float*)d_in[15];
  float* out = (float*)d_out;
  float* mp  = (float*)d_ws;   // 96*158 f32 scratch

  mpnn_scan_kernel<<<M_TOT / BM, THREADS, LDS_BYTES, stream>>>(
      a2b, b2a, b2revb, f_bonds, bond_sum, W0, b0, mp);
  mlp_head_kernel<<<M_TOT, 128, 0, stream>>>(mp, W1, b1, W2, b2, W3, b3, W4, b4, out);
}
